// ExplicitSVDBlock_85529978733213
// MI455X (gfx1250) — matrix operand order repaired and checked by reading, not compiled, not run on hardware
//
#include <hip/hip_runtime.h>
#include <hip/hip_bf16.h>

typedef __attribute__((ext_vector_type(16))) _Float16 v16h;
typedef __attribute__((ext_vector_type(8)))  float    v8f;
typedef __attribute__((ext_vector_type(4)))  int      v4i;

#define DEV static __device__ __forceinline__

constexpr int CB   = 4;       // batch
constexpr int CS   = 2048;    // seq
constexpr int CD   = 768;     // model dim
constexpr int CH   = 12;      // heads
constexpr int CHD  = 64;      // head dim
constexpr int CRA  = 32;      // attn rank
constexpr int CRF  = 512;     // ffn rank
constexpr int CDFF = 3072;    // ffn dim
constexpr int CBS  = CB * CS; // 8192 tokens

// ---------------- CDNA5 async global->LDS staging (guarded) ----------------
#if defined(__HIP_DEVICE_COMPILE__) && defined(__gfx1250__) &&                 \
    __has_builtin(__builtin_amdgcn_global_load_async_to_lds_b128) &&           \
    __has_builtin(__builtin_amdgcn_s_wait_asynccnt)
#define HAVE_ASYNC_LDS 1
#else
#define HAVE_ASYNC_LDS 0
#endif

// copy 16 bytes global -> LDS (async on gfx1250 when available)
DEV void stage16(const _Float16* g, _Float16* l) {
#if HAVE_ASYNC_LDS
  __builtin_amdgcn_global_load_async_to_lds_b128((v4i*)g, (v4i*)l, 0, 0);
#else
  *(uint4*)l = *(const uint4*)g;
#endif
}

// wait until at most N async copies remain in flight (in-order completion
// means everything issued before the newest N is then resident in LDS)
template <int N> DEV void stage_waitn() {
#if HAVE_ASYNC_LDS
  __builtin_amdgcn_s_wait_asynccnt(N);
#endif
}

// ---------------- WMMA fragment helpers (contiguous k order, k = 16*(lane/16) + s) ----------------

DEV v8f wmma_f32_f16(v16h a, v16h b, v8f c) {
#if defined(__gfx1250__)
  return __builtin_amdgcn_wmma_f32_16x16x32_f16(false, a, false, b, (short)0, c,
                                                false, false);
#else
  (void)a; (void)b;
  return c;
#endif
}

// A fragment: 16x32 f16, row M = lane&15; lanes 0-15 hold K 0..15,
// lanes 16-31 hold K 16..31. Two 16B vector loads per lane.
DEV v16h load_a_f16(const _Float16* A, int lda) {
  int lane = threadIdx.x & 31;
  int row  = lane & 15;
  int kb   = (lane & 16) ? 16 : 0;
  const _Float16* p = A + (size_t)row * lda + kb;
  union { uint4 q[2]; v16h v; } u;
  u.q[0] = *(const uint4*)(p);
  u.q[1] = *(const uint4*)(p + 8);
  return u.v;
}

// B fragment from N-major (transposed) storage: logical B(k,n) = mem[n*ldb + k].
// col N = lane&15; lanes 0-15 hold K 0..15, lanes 16-31 hold K 16..31.
DEV v16h load_bt_f16(const _Float16* B, int ldb) {
  int lane = threadIdx.x & 31;
  int col  = lane & 15;
  int kb   = (lane & 16) ? 16 : 0;
  const _Float16* p = B + (size_t)col * ldb + kb;
  union { uint4 q[2]; v16h v; } u;
  u.q[0] = *(const uint4*)(p);
  u.q[1] = *(const uint4*)(p + 8);
  return u.v;
}

// B fragment from K-major storage (used for V in attention): B(k,n)=mem[k*ldb+n]
DEV v16h load_b_f16(const _Float16* B, int ldb) {
  int lane = threadIdx.x & 31;
  int col  = lane & 15;
  int kb   = (lane & 16) ? 16 : 0;
  v16h b;
#pragma unroll
  for (int i = 0; i < 16; ++i) b[i] = B[(size_t)(kb + i) * ldb + col];
  return b;
}

DEV float gelu_tanh(float x) {
  float x3 = x * x * x;
  return 0.5f * x * (1.0f + tanhf(0.7978845608028654f * (x + 0.044715f * x3)));
}

// ---------------- Weight conversion: f32 [K,N] -> f16 [N,K] (transpose) ----------------
__global__ void cvt_t_kernel(const float* __restrict__ in,
                             _Float16* __restrict__ out, int K, int N) {
  __shared__ float tile[32][33];
  in  += (size_t)blockIdx.z * K * N;
  out += (size_t)blockIdx.z * K * N;
  int k0 = blockIdx.x * 32, n0 = blockIdx.y * 32;
  int tx = threadIdx.x & 31, ty = threadIdx.x >> 5;  // 32 x 8
#pragma unroll
  for (int i = ty; i < 32; i += 8)
    tile[i][tx] = in[(size_t)(k0 + i) * N + n0 + tx];
  __syncthreads();
#pragma unroll
  for (int i = ty; i < 32; i += 8)
    out[(size_t)(n0 + i) * K + k0 + tx] = (_Float16)tile[tx][i];
}

// straight f32 -> f16 (layout already [N,K])
__global__ void cvt_kernel(const float* __restrict__ in,
                           _Float16* __restrict__ out, int n) {
  int i = blockIdx.x * 256 + threadIdx.x;
  if (i < n) out[i] = (_Float16)in[i];
}

// ---------------- LayerNorm (f32 in) -> f16 out ----------------
__global__ void ln_f16_kernel(const float* __restrict__ x,
                              const float* __restrict__ w,
                              const float* __restrict__ b,
                              _Float16* __restrict__ out) {
  int row = blockIdx.x;
  int tid = threadIdx.x;
  const float* xr = x + (size_t)row * CD;
  float v0 = xr[tid], v1 = xr[tid + 256], v2 = xr[tid + 512];
  float s  = v0 + v1 + v2;
  float sq = v0 * v0 + v1 * v1 + v2 * v2;
#pragma unroll
  for (int m = 16; m >= 1; m >>= 1) {
    s  += __shfl_xor(s, m, 32);
    sq += __shfl_xor(sq, m, 32);
  }
  __shared__ float ws_[8], wq_[8];
  int lane = tid & 31, wv = tid >> 5;
  if (lane == 0) { ws_[wv] = s; wq_[wv] = sq; }
  __syncthreads();
  s = 0.f; sq = 0.f;
#pragma unroll
  for (int i = 0; i < 8; ++i) { s += ws_[i]; sq += wq_[i]; }
  float mean = s * (1.0f / CD);
  float var  = sq * (1.0f / CD) - mean * mean;
  float rstd = rsqrtf(var + 1e-6f);
  _Float16* o = out + (size_t)row * CD;
  o[tid]       = (_Float16)((v0 - mean) * rstd * w[tid]       + b[tid]);
  o[tid + 256] = (_Float16)((v1 - mean) * rstd * w[tid + 256] + b[tid + 256]);
  o[tid + 512] = (_Float16)((v2 - mean) * rstd * w[tid + 512] + b[tid + 512]);
}

// ---------------- Per-head low-rank projection (+bias, optional RoPE) ----------------
// h16:[BS,D] f16 ; Ut:[H,RA,D] f16 ; Vt:[H,HD,RA] f16 ; bias:[D] f32 ; out:[B,H,S,HD] f16
__global__ void qkv_kernel(const _Float16* __restrict__ h16,
                           const _Float16* __restrict__ Ut,
                           const _Float16* __restrict__ Vt,
                           const float* __restrict__ bias,
                           const float* __restrict__ cosT,
                           const float* __restrict__ sinT,
                           _Float16* __restrict__ outQ) {
  __shared__ __align__(16) _Float16 ldsU[2][32 * 32];  // U tile: [n=RA][k=32], dbl buf
  __shared__ __align__(16) _Float16 ldsVw[64 * 32];    // V wts : [n=HD][k=RA]
  __shared__ __align__(16) _Float16 ldsXU[4][16 * CRA];
  int wv = threadIdx.x >> 5, lane = threadIdx.x & 31;
  int col = lane & 15, rb = (lane & 16) ? 8 : 0;
  int h  = blockIdx.y;
  int m0 = blockIdx.x * 64 + wv * 16;
  const _Float16* Uh = Ut + (size_t)h * CRA * CD;
  const _Float16* Vh = Vt + (size_t)h * CHD * CRA;

  auto stageU = [&](int buf, int kk) {  // 1 async b128 per thread
    int r = threadIdx.x >> 2, c = (threadIdx.x & 3) * 8;
    stage16(Uh + (size_t)r * CD + kk + c, ldsU[buf] + r * 32 + c);
  };

  // stage V weights once (2 copies/thread) and the first U tile
  for (int g = threadIdx.x; g < 256; g += 128)
    stage16(Vh + g * 8, ldsVw + g * 8);
  stageU(0, 0);

  // xu = h @ U[h]  (16 x 32)
  v8f acc[2] = {};
  constexpr int NIT = CD / 32;
  for (int it = 0; it < NIT; ++it) {
    int kk = it * 32, cur = it & 1;
    if (it + 1 < NIT) {
      stageU(cur ^ 1, kk + 32);   // overlap next-tile copy with compute
      stage_waitn<1>();           // older copies (incl. current tile) resident
      __builtin_prefetch(h16 + (size_t)m0 * CD + kk + 32 + (threadIdx.x & 15) * 48, 0, 3);
    } else {
      stage_waitn<0>();
    }
    __syncthreads();
    v16h a = load_a_f16(h16 + (size_t)m0 * CD + kk, CD);
#pragma unroll
    for (int nt = 0; nt < 2; ++nt) {
      v16h bf = load_bt_f16(ldsU[cur] + nt * 16 * 32, 32);
      acc[nt] = wmma_f32_f16(a, bf, acc[nt]);
    }
    __syncthreads();
  }
  // restage xu C-layout -> row-major f16 (wave-private, in-order DS)
#pragma unroll
  for (int nt = 0; nt < 2; ++nt)
#pragma unroll
    for (int r = 0; r < 8; ++r)
      ldsXU[wv][(rb + r) * CRA + nt * 16 + col] = (_Float16)acc[nt][r];

  v16h axu = load_a_f16(ldsXU[wv], CRA);
  v8f q[4] = {};
#pragma unroll
  for (int nt = 0; nt < 4; ++nt) {
    v16h bf = load_bt_f16(ldsVw + nt * 16 * 32, 32);
    q[nt] = wmma_f32_f16(axu, bf, q[nt]);
  }

  int bIdx = m0 / CS;
  int s0   = m0 % CS;
  float qv[4][8], res[4][8];
#pragma unroll
  for (int nt = 0; nt < 4; ++nt) {
    float bv = bias[h * CHD + nt * 16 + col];
#pragma unroll
    for (int r = 0; r < 8; ++r) qv[nt][r] = q[nt][r] + bv;
  }
  if (cosT) {
#pragma unroll
    for (int nt = 0; nt < 4; ++nt) {
      int d = nt * 16 + col;
#pragma unroll
      for (int r = 0; r < 8; ++r) {
        int srow = s0 + rb + r;
        float c  = cosT[(size_t)srow * CHD + d];
        float sn = sinT[(size_t)srow * CHD + d];
        float other = qv[nt ^ 2][r];               // element at d +/- 32
        float rh = (d < 32) ? -other : other;      // rotate_half
        res[nt][r] = qv[nt][r] * c + rh * sn;
      }
    }
  } else {
#pragma unroll
    for (int nt = 0; nt < 4; ++nt)
#pragma unroll
      for (int r = 0; r < 8; ++r) res[nt][r] = qv[nt][r];
  }
  _Float16* op = outQ + (((size_t)bIdx * CH + h) * CS + s0) * CHD;
#pragma unroll
  for (int nt = 0; nt < 4; ++nt)
#pragma unroll
    for (int r = 0; r < 8; ++r)
      op[(size_t)(rb + r) * CHD + nt * 16 + col] = (_Float16)res[nt][r];
}

// ---------------- Flash attention per (b,h), wave = 16 query rows ----------------
__global__ void attn_kernel(const _Float16* __restrict__ qg,
                            const _Float16* __restrict__ kg,
                            const _Float16* __restrict__ vg,
                            _Float16* __restrict__ og) {
  __shared__ __align__(16) _Float16 ldsK[2][32 * CHD];
  __shared__ __align__(16) _Float16 ldsV[2][32 * CHD];
  __shared__ __align__(16) _Float16 ldsP[4][16 * 32];
  __shared__ float lm[4][16], ll[4][16];

  int wv = threadIdx.x >> 5, lane = threadIdx.x & 31;
  int col = lane & 15, rb = (lane & 16) ? 8 : 0;
  int bh = blockIdx.y;
  int q0 = blockIdx.x * 64 + wv * 16;
  const _Float16* Qp = qg + ((size_t)bh * CS + q0) * CHD;
  const _Float16* Kp = kg + (size_t)bh * CS * CHD;
  const _Float16* Vp = vg + (size_t)bh * CS * CHD;

  v16h aq0 = load_a_f16(Qp + 0, CHD);
  v16h aq1 = load_a_f16(Qp + 32, CHD);

  if (lane < 16) { lm[wv][lane] = -1e30f; ll[wv][lane] = 0.f; }

  v8f accO[4] = {};
  const float scale = 0.125f; // 1/sqrt(64)

  auto stageKV = [&](int buf, int kv) {  // 4 async b128 per thread
    const _Float16* gk = Kp + (size_t)kv * CHD;
    const _Float16* gv = Vp + (size_t)kv * CHD;
    for (int g = threadIdx.x; g < 256; g += 128) {
      stage16(gk + g * 8, ldsK[buf] + g * 8);
      stage16(gv + g * 8, ldsV[buf] + g * 8);
    }
  };
  stageKV(0, 0);

  constexpr int NKV = CS / 32;
  for (int it = 0; it < NKV; ++it) {
    int cur = it & 1;
    if (it + 1 < NKV) {
      stageKV(cur ^ 1, (it + 1) * 32);  // overlap next chunk with compute
      stage_waitn<4>();
    } else {
      stage_waitn<0>();
    }
    __syncthreads();

    // scores [16 x 32] = Q @ K^T (K-dim = HD = 64 -> 2 wmma per tile)
    v8f sc[2];
#pragma unroll
    for (int nt = 0; nt < 2; ++nt) {
      v16h b0 = load_bt_f16(ldsK[cur] + (nt * 16) * CHD + 0, CHD);
      v16h b1 = load_bt_f16(ldsK[cur] + (nt * 16) * CHD + 32, CHD);
      v8f t = {};
      t = wmma_f32_f16(aq0, b0, t);
      t = wmma_f32_f16(aq1, b1, t);
      sc[nt] = t;
    }
#pragma unroll
    for (int nt = 0; nt < 2; ++nt)
#pragma unroll
      for (int r = 0; r < 8; ++r) sc[nt][r] = sc[nt][r] * scale;

    // online softmax: per-row stats (row = rb + r, spread over 16 lanes)
    float alpha[8];
#pragma unroll
    for (int r = 0; r < 8; ++r) {
      float mx = fmaxf(sc[0][r], sc[1][r]);
#pragma unroll
      for (int m = 1; m <= 8; m <<= 1) mx = fmaxf(mx, __shfl_xor(mx, m, 32));
      float mo = lm[wv][rb + r];
      float mn = fmaxf(mo, mx);
      float al = __expf(mo - mn);
      alpha[r] = al;
      float p0 = __expf(sc[0][r] - mn);
      float p1 = __expf(sc[1][r] - mn);
      sc[0][r] = p0;
      sc[1][r] = p1;
      float rs = p0 + p1;
#pragma unroll
      for (int m = 1; m <= 8; m <<= 1) rs += __shfl_xor(rs, m, 32);
      if (col == 0) {
        ll[wv][rb + r] = al * ll[wv][rb + r] + rs;
        lm[wv][rb + r] = mn;
      }
    }
    // rescale O accumulator
#pragma unroll
    for (int dt = 0; dt < 4; ++dt)
#pragma unroll
      for (int r = 0; r < 8; ++r) accO[dt][r] = accO[dt][r] * alpha[r];

    // stage P (C-layout) -> row-major f16, reload as A fragment
#pragma unroll
    for (int nt = 0; nt < 2; ++nt)
#pragma unroll
      for (int r = 0; r < 8; ++r)
        ldsP[wv][(rb + r) * 32 + nt * 16 + col] = (_Float16)sc[nt][r];
    v16h ap = load_a_f16(ldsP[wv], 32);
#pragma unroll
    for (int dt = 0; dt < 4; ++dt) {
      v16h bv = load_b_f16(ldsV[cur] + dt * 16, CHD);
      accO[dt] = wmma_f32_f16(ap, bv, accO[dt]);
    }
    __syncthreads();
  }

  int bIdx = bh / CH, h = bh % CH;
  _Float16* op = og + ((size_t)bIdx * CS + q0) * CD + h * CHD;
#pragma unroll
  for (int dt = 0; dt < 4; ++dt)
#pragma unroll
    for (int r = 0; r < 8; ++r) {
      float denom = ll[wv][rb + r];
      op[(size_t)(rb + r) * CD + dt * 16 + col] = (_Float16)(accO[dt][r] / denom);
    }
}

// -------- Generic WMMA GEMM: out = A(f16 [M,K]) @ Bt(f16 [N,K])^T [+bias][+res] --------
// Block = 4 waves; each wave owns a 32x64 output tile (2 A frags x 4 B frags
// = 8 WMMA per k-step, reusing each LDS B fragment twice). Double-buffered
// async B staging overlaps copies with compute.
template <bool OUT16, bool HASBIAS, bool HASRES>
__global__ void gemm_kernel(const _Float16* __restrict__ A,
                            const _Float16* __restrict__ Bt,
                            const float* __restrict__ bias,
                            const float* __restrict__ res,
                            void* __restrict__ outv,
                            int M, int N, int K) {
  __shared__ __align__(16) _Float16 ldsB[2][64 * 32];  // [n_local][k], dbl buf
  int wv = threadIdx.x >> 5, lane = threadIdx.x & 31;
  int col = lane & 15, rb = (lane & 16) ? 8 : 0;
  int m0 = blockIdx.x * 128 + wv * 32;
  int n0 = blockIdx.y * 64;
  (void)M;

  auto stageB = [&](int buf, int kk) {  // 2 async b128 per thread
    for (int g = threadIdx.x; g < 256; g += 128) {
      int r = g >> 2, c = (g & 3) * 8;
      stage16(Bt + (size_t)(n0 + r) * K + kk + c, ldsB[buf] + r * 32 + c);
    }
  };
  stageB(0, 0);

  v8f acc0[4] = {}, acc1[4] = {};
  int nit = K / 32;
  for (int it = 0; it < nit; ++it) {
    int kk = it * 32, cur = it & 1;
    if (it + 1 < nit) {
      stageB(cur ^ 1, kk + 32);
      stage_waitn<2>();
      __builtin_prefetch(A + (size_t)(m0 + (threadIdx.x & 31)) * K + kk + 32, 0, 3);
    } else {
      stage_waitn<0>();
    }
    __syncthreads();
    v16h a0 = load_a_f16(A + (size_t)m0 * K + kk, K);
    v16h a1 = load_a_f16(A + (size_t)(m0 + 16) * K + kk, K);
#pragma unroll
    for (int nt = 0; nt < 4; ++nt) {
      v16h b = load_bt_f16(ldsB[cur] + nt * 16 * 32, 32);
      acc0[nt] = wmma_f32_f16(a0, b, acc0[nt]);
      acc1[nt] = wmma_f32_f16(a1, b, acc1[nt]);
    }
    __syncthreads();
  }
#pragma unroll
  for (int half = 0; half < 2; ++half) {
    int mb = m0 + half * 16;
#pragma unroll
    for (int nt = 0; nt < 4; ++nt) {
      int n = n0 + nt * 16 + col;
      float bv = HASBIAS ? bias[n] : 0.f;
#pragma unroll
      for (int r = 0; r < 8; ++r) {
        int row = mb + rb + r;
        float v = (half ? acc1[nt][r] : acc0[nt][r]) + bv;
        if constexpr (HASRES) v += res[(size_t)row * N + n];
        if constexpr (OUT16)
          ((_Float16*)outv)[(size_t)row * N + n] = (_Float16)v;
        else
          ((float*)outv)[(size_t)row * N + n] = v;
      }
    }
  }
}

// ---------------- Fused GEGLU: g = gelu(t1@Vi_g+bi_g) * (t1@Vi_v+bi_v) ----------------
// A:[BS,RF] f16 ; ViT:[2*DFF, RF] f16 (N-major) ; bi:[2*DFF] f32 ; out:[BS,DFF] f16
__global__ void geglu_kernel(const _Float16* __restrict__ A,
                             const _Float16* __restrict__ ViT,
                             const float* __restrict__ bi,
                             _Float16* __restrict__ out) {
  __shared__ __align__(16) _Float16 ldsB1[2][64 * 32];
  __shared__ __align__(16) _Float16 ldsB2[2][64 * 32];
  int wv = threadIdx.x >> 5, lane = threadIdx.x & 31;
  int col = lane & 15, rb = (lane & 16) ? 8 : 0;
  int m0 = blockIdx.x * 64 + wv * 16;
  int n0 = blockIdx.y * 64;
  const _Float16* B1 = ViT + (size_t)n0 * CRF;
  const _Float16* B2 = ViT + (size_t)(n0 + CDFF) * CRF;

  auto stageB = [&](int buf, int kk) {  // 4 async b128 per thread
    for (int g = threadIdx.x; g < 256; g += 128) {
      int r = g >> 2, c = (g & 3) * 8;
      stage16(B1 + (size_t)r * CRF + kk + c, ldsB1[buf] + r * 32 + c);
      stage16(B2 + (size_t)r * CRF + kk + c, ldsB2[buf] + r * 32 + c);
    }
  };
  stageB(0, 0);

  v8f a1[4] = {}, a2[4] = {};
  constexpr int NIT = CRF / 32;
  for (int it = 0; it < NIT; ++it) {
    int kk = it * 32, cur = it & 1;
    if (it + 1 < NIT) {
      stageB(cur ^ 1, kk + 32);
      stage_waitn<4>();
      __builtin_prefetch(A + (size_t)(m0 + (threadIdx.x & 15)) * CRF + kk + 32, 0, 3);
    } else {
      stage_waitn<0>();
    }
    __syncthreads();
    v16h a = load_a_f16(A + (size_t)m0 * CRF + kk, CRF);
#pragma unroll
    for (int nt = 0; nt < 4; ++nt) {
      v16h b1 = load_bt_f16(ldsB1[cur] + nt * 16 * 32, 32);
      v16h b2 = load_bt_f16(ldsB2[cur] + nt * 16 * 32, 32);
      a1[nt] = wmma_f32_f16(a, b1, a1[nt]);
      a2[nt] = wmma_f32_f16(a, b2, a2[nt]);
    }
    __syncthreads();
  }
#pragma unroll
  for (int nt = 0; nt < 4; ++nt) {
    int n = n0 + nt * 16 + col;
    float b1 = bi[n], b2 = bi[n + CDFF];
#pragma unroll
    for (int r = 0; r < 8; ++r) {
      int row = m0 + rb + r;
      float g1 = a1[nt][r] + b1;
      float g2 = a2[nt][r] + b2;
      out[(size_t)row * CDFF + n] = (_Float16)(gelu_tanh(g1) * g2);
    }
  }
}

// ---------------- Host launch ----------------
extern "C" void kernel_launch(void* const* d_in, const int* in_sizes, int n_in,
                              void* d_out, int out_size, void* d_ws, size_t ws_size,
                              hipStream_t stream) {
  (void)in_sizes; (void)n_in; (void)out_size; (void)ws_size;
  const float* x     = (const float*)d_in[0];
  const float* ln1_w = (const float*)d_in[1];
  const float* ln1_b = (const float*)d_in[2];
  const float* Uq    = (const float*)d_in[3];
  const float* Vq    = (const float*)d_in[4];
  const float* bq    = (const float*)d_in[5];
  const float* Uk    = (const float*)d_in[6];
  const float* Vk    = (const float*)d_in[7];
  const float* bk    = (const float*)d_in[8];
  const float* Uv    = (const float*)d_in[9];
  const float* Vv    = (const float*)d_in[10];
  const float* bv    = (const float*)d_in[11];
  const float* Wo_w  = (const float*)d_in[12];
  const float* Wo_b  = (const float*)d_in[13];
  const float* ln2_w = (const float*)d_in[14];
  const float* ln2_b = (const float*)d_in[15];
  const float* Ui    = (const float*)d_in[16];
  const float* Vi    = (const float*)d_in[17];
  const float* bi    = (const float*)d_in[18];
  const float* Uo    = (const float*)d_in[19];
  const float* Vo    = (const float*)d_in[20];
  const float* bo    = (const float*)d_in[21];
  const float* cosT  = (const float*)d_in[22];
  const float* sinT  = (const float*)d_in[23];

  char* wsp = (char*)d_ws;
  auto take = [&](size_t bytes) {
    char* p = wsp;
    wsp += (bytes + 255) & ~(size_t)255;
    return p;
  };
  // activations
  _Float16* h16 = (_Float16*)take((size_t)CBS * CD * 2);
  _Float16* q16 = (_Float16*)take((size_t)CBS * CD * 2);
  _Float16* k16 = (_Float16*)take((size_t)CBS * CD * 2);
  _Float16* v16 = (_Float16*)take((size_t)CBS * CD * 2);
  _Float16* o16 = (_Float16*)take((size_t)CBS * CD * 2);
  float*    x1  = (float*)   take((size_t)CBS * CD * 4);
  _Float16* h2  = (_Float16*)take((size_t)CBS * CD * 2);
  _Float16* t1  = (_Float16*)take((size_t)CBS * CRF * 2);
  _Float16* g16 = (_Float16*)take((size_t)CBS * CDFF * 2);
  _Float16* t2  = (_Float16*)take((size_t)CBS * CRF * 2);
  // f16 weights, N-major [N,K]
  _Float16* UqT = (_Float16*)take((size_t)CH * CRA * CD * 2);
  _Float16* UkT = (_Float16*)take((size_t)CH * CRA * CD * 2);
  _Float16* UvT = (_Float16*)take((size_t)CH * CRA * CD * 2);
  _Float16* VqT = (_Float16*)take((size_t)CH * CHD * CRA * 2);
  _Float16* VkT = (_Float16*)take((size_t)CH * CHD * CRA * 2);
  _Float16* VvT = (_Float16*)take((size_t)CH * CHD * CRA * 2);
  _Float16* Wo16 = (_Float16*)take((size_t)CD * CD * 2);
  _Float16* UiT  = (_Float16*)take((size_t)CRF * CD * 2);
  _Float16* ViT  = (_Float16*)take((size_t)2 * CDFF * CRF * 2);
  _Float16* UoT  = (_Float16*)take((size_t)CRF * CDFF * 2);
  _Float16* VoT  = (_Float16*)take((size_t)CD * CRF * 2);

  auto cvt_t = [&](const float* in, _Float16* out, int K, int N, int batch) {
    cvt_t_kernel<<<dim3(K / 32, N / 32, batch), 256, 0, stream>>>(in, out, K, N);
  };
  // 0. weight precision/layout conversion
  cvt_t(Uq, UqT, CD, CRA, CH);    // [H,D,RA] -> [H,RA,D]
  cvt_t(Uk, UkT, CD, CRA, CH);
  cvt_t(Uv, UvT, CD, CRA, CH);
  cvt_t(Vq, VqT, CRA, CHD, CH);   // [H,RA,HD] -> [H,HD,RA]
  cvt_t(Vk, VkT, CRA, CHD, CH);
  cvt_t(Vv, VvT, CRA, CHD, CH);
  cvt_kernel<<<(CD * CD + 255) / 256, 256, 0, stream>>>(Wo_w, Wo16, CD * CD);
  cvt_t(Ui, UiT, CD, CRF, 1);          // [D,RF]      -> [RF,D]
  cvt_t(Vi, ViT, CRF, 2 * CDFF, 1);    // [RF,2DFF]   -> [2DFF,RF]
  cvt_t(Uo, UoT, CDFF, CRF, 1);        // [DFF,RF]    -> [RF,DFF]
  cvt_t(Vo, VoT, CRF, CD, 1);          // [RF,D]      -> [D,RF]

  // 1. pre-norm
  ln_f16_kernel<<<CBS, 256, 0, stream>>>(x, ln1_w, ln1_b, h16);

  // 2. q/k/v low-rank projections (+RoPE on q,k)
  dim3 qg(CBS / 64, CH);
  qkv_kernel<<<qg, 128, 0, stream>>>(h16, UqT, VqT, bq, cosT, sinT, q16);
  qkv_kernel<<<qg, 128, 0, stream>>>(h16, UkT, VkT, bk, cosT, sinT, k16);
  qkv_kernel<<<qg, 128, 0, stream>>>(h16, UvT, VvT, bv, nullptr, nullptr, v16);

  // 3. attention
  attn_kernel<<<dim3(CS / 64, CB * CH), 128, 0, stream>>>(q16, k16, v16, o16);

  // 4. x1 = x + o @ Wo^T + Wo_b   (Wo16 already [N,K])
  gemm_kernel<false, true, true>
      <<<dim3(CBS / 128, CD / 64), 128, 0, stream>>>(o16, Wo16, Wo_b, x, x1,
                                                     CBS, CD, CD);

  // 5. mlp pre-norm
  ln_f16_kernel<<<CBS, 256, 0, stream>>>(x1, ln2_w, ln2_b, h2);

  // 6. t1 = h2 @ Ui
  gemm_kernel<true, false, false>
      <<<dim3(CBS / 128, CRF / 64), 128, 0, stream>>>(h2, UiT, nullptr, nullptr,
                                                      t1, CBS, CRF, CD);

  // 7. g = gelu(t1@Vi_g + bi_g) * (t1@Vi_v + bi_v)
  geglu_kernel<<<dim3(CBS / 64, CDFF / 64), 128, 0, stream>>>(t1, ViT, bi, g16);

  // 8. t2 = g @ Uo
  gemm_kernel<true, false, false>
      <<<dim3(CBS / 128, CRF / 64), 128, 0, stream>>>(g16, UoT, nullptr, nullptr,
                                                      t2, CBS, CRF, CDFF);

  // 9. out = x1 + t2 @ Vo + bo
  gemm_kernel<false, true, true>
      <<<dim3(CBS / 128, CD / 64), 128, 0, stream>>>(t2, VoT, bo, x1,
                                                     (float*)d_out, CBS, CD, CRF);
}
